// DeepComplexIrisNet_37340445672204
// MI455X (gfx1250) — matrix operand order repaired and checked by reading, not compiled, not run on hardware
//
#include <hip/hip_runtime.h>
#include <hip/hip_bf16.h>
#include <math.h>

// ---------------------------------------------------------------------------
// DeepComplexIrisNet forward on MI455X (gfx1250, wave32, WMMA).
// Every conv / FC layer is an implicit GEMM on v_wmma_f32_16x16x32_f16
// (f16 operands, f32 accumulate). Complex conv folds into one real GEMM with
// the block weight matrix [[Wr,-Wi],[Wi,Wr]]. BN(whitening)+CReLU+|z|-maxpool
// is fused into a single streaming pass per layer.
// N-blocking: each block stages NB=4 B-panels; each wave runs 4 independent
// WMMA accumulators off one shared A fragment (amortizes A traffic, hides
// XDL latency).
// ---------------------------------------------------------------------------

typedef _Float16 h16;
typedef __attribute__((ext_vector_type(16))) _Float16 v16h;
typedef __attribute__((ext_vector_type(8)))  _Float16 v8h;
typedef __attribute__((ext_vector_type(8)))  float    v8f;

#define BN_EPS 1e-5f
#define MAG_EPS 1e-12f
#define NB 4                       // N-tiles (16-col panels) per block / per wave

// ------------------------------- prep kernels ------------------------------

__global__ void k_cast_f16(const float* __restrict__ in, h16* __restrict__ out, int n) {
  int i = blockIdx.x * blockDim.x + threadIdx.x;
  if (i < n) out[i] = (h16)in[i];
}

// conv1: input imag == 0 -> two real convs. Wp [32][128] (K=121 zero-padded).
__global__ void k_prep_conv1(const float* __restrict__ wr, const float* __restrict__ wi,
                             const float* __restrict__ br, const float* __restrict__ bi,
                             h16* __restrict__ Wp, float* __restrict__ bias) {
  int i = blockIdx.x * blockDim.x + threadIdx.x;
  if (i >= 32 * 128) return;
  int m = i >> 7, k = i & 127;
  float v = 0.f;
  if (k < 121) v = (m < 16) ? wr[m * 121 + k] : wi[(m - 16) * 121 + k];
  Wp[i] = (h16)v;
  if (k == 0) bias[m] = (m < 16) ? (br[m] - bi[m]) : (br[m - 16] + bi[m - 16]);
}

// complex conv: Wp [2co][2*ci*kk] = [[Wr,-Wi],[Wi,Wr]]; bias = br-bi | br+bi
__global__ void k_prep_convc(const float* __restrict__ wr, const float* __restrict__ wi,
                             const float* __restrict__ br, const float* __restrict__ bi,
                             h16* __restrict__ Wp, float* __restrict__ bias,
                             int co, int cikk) {
  int K = 2 * cikk;
  int tot = 2 * co * K;
  int i = blockIdx.x * blockDim.x + threadIdx.x;
  if (i >= tot) return;
  int m = i / K, k = i - m * K;
  float v;
  if (m < co) v = (k < cikk) ? wr[m * cikk + k] : -wi[m * cikk + (k - cikk)];
  else        v = (k < cikk) ? wi[(m - co) * cikk + k] : wr[(m - co) * cikk + (k - cikk)];
  Wp[i] = (h16)v;
  if (k == 0) bias[m] = (m < co) ? (br[m] - bi[m]) : (br[m - co] + bi[m - co]);
}

// fc1 complex weight block: Wc [512][2048]
__global__ void k_prep_fc1(const float* __restrict__ wr, const float* __restrict__ wi,
                           h16* __restrict__ Wc) {
  int i = blockIdx.x * blockDim.x + threadIdx.x;
  if (i >= 512 * 2048) return;
  int j = i >> 11, k = i & 2047;
  float v;
  if (j < 256) v = (k < 1024) ? wr[j * 1024 + k] : -wi[j * 1024 + (k - 1024)];
  else         v = (k < 1024) ? wi[(j - 256) * 1024 + k] : wr[(j - 256) * 1024 + (k - 1024)];
  Wc[i] = (h16)v;
}

// ------------------------- implicit-GEMM conv (WMMA) -----------------------
// C[Mrows x Npix] = Wp[Mrows x Kp] * im2col(act)[Kp x Npix]
// act layout  : [N][inBlocks*Cin][H][W] f16 (real channels then imag channels)
// out layout  : [N][Mrows][H][W] f32 (rows<Mrows/2 real, rest imag)
// Block covers NB 16-col panels; wave w covers M tile w with NB accumulators.
__global__ void k_conv_wmma(const h16* __restrict__ Wp, const float* __restrict__ bias,
                            const h16* __restrict__ act, float* __restrict__ out,
                            int N, int Cin, int inBlocks, int H, int W,
                            int KH, int KW, int pad, int Mrows, int K, int Kp, int KC) {
  extern __shared__ v16h ldsraw[];
  h16* lds = (h16*)ldsraw;
  const int KCpad = KC + 16;                 // multiple of 16 -> 32B aligned rows
  const int COLS = 16 * NB;
  const int HW = H * W;
  const int KK = KH * KW;
  const int CinKK = Cin * KK;
  const int tid = threadIdx.x;
  const int lane = tid & 31;
  const int wave = tid >> 5;
  const int col = lane & 15;
  const int khalf = lane >> 4;
  const int mTile = wave;                    // one wave per M tile (Mrows/16 waves)
  const int Mtiles = Mrows >> 4;
  const int pBase = blockIdx.x * COLS;

  v8f acc[NB] = {};

  for (int kc0 = 0; kc0 < Kp; kc0 += KC) {
    // ---- cooperative im2col staging of NB B-panels: lds[colG][kLoc] ----
    for (int i = tid; i < COLS * KC; i += blockDim.x) {
      int colG = i / KC;
      int kLoc = i - colG * KC;
      int k = kc0 + kLoc;
      float v = 0.f;
      if (k < K) {
        int pp = pBase + colG;
        int nn = pp / HW;
        int rpp = pp - nn * HW;
        int oyy = rpp / W;
        int oxx = rpp - oyy * W;
        int blk = k / CinKK;
        int rem = k - blk * CinKK;
        int ci  = rem / KK;
        int r   = rem - ci * KK;
        int ky  = r / KW;
        int kx  = r - ky * KW;
        int iy = oyy + ky - pad, ix = oxx + kx - pad;
        if (iy >= 0 && iy < H && ix >= 0 && ix < W)
          v = (float)act[(((long)nn * (inBlocks * Cin) + blk * Cin + ci) * H + iy) * W + ix];
      }
      lds[colG * KCpad + kLoc] = (h16)v;
    }
    __syncthreads();

    if (mTile < Mtiles) {
      int kcLen = Kp - kc0; if (kcLen > KC) kcLen = KC;
      // A row for this lane: M = mTile*16 + (lane&15); K-halves split by lane half
      const h16* Arow = Wp + (long)(mTile * 16 + col) * Kp + kc0 + 16 * khalf;
      const h16* Bcol = lds + col * KCpad + 16 * khalf;
      for (int k0 = 0; k0 < kcLen; k0 += 32) {
        v8h alo = *(const v8h*)(Arow + k0);        // K = k0 + 16*kh + {0..7}
        v8h ahi = *(const v8h*)(Arow + k0 + 8);    // K = k0 + 16*kh + 8 + {0..7}
        v16h a;
#pragma unroll
        for (int t = 0; t < 8; ++t) { a[t] = alo[t]; a[8 + t] = ahi[t]; }
#pragma unroll
        for (int nb = 0; nb < NB; ++nb) {          // 4 independent accumulators
          v16h b = *(const v16h*)(Bcol + nb * 16 * KCpad + k0);
          acc[nb] = __builtin_amdgcn_wmma_f32_16x16x32_f16(
              false, a, false, b, (short)0, acc[nb], false, false);
        }
      }
    }
    __syncthreads();
  }

  if (mTile < Mtiles) {
#pragma unroll
    for (int nb = 0; nb < NB; ++nb) {
      int p = pBase + nb * 16 + col;
      int n = p / HW;
      int rp = p - n * HW;
#pragma unroll
      for (int v = 0; v < 8; ++v) {
        int row = mTile * 16 + v + 8 * khalf;      // C layout: VGPR v -> M = v + 8*kh
        out[((long)n * Mrows + row) * HW + rp] = acc[nb][v] + bias[row];
      }
    }
  }
}

// ------------------- complex BN stats -> folded 2x2 affine ------------------
// One block per channel. conv layout [N][2C][H][W] (real c, imag C+c).
__global__ void k_bn_stats(const float* __restrict__ conv, int N, int C, int HW,
                           const float* __restrict__ grr, const float* __restrict__ gri,
                           const float* __restrict__ gii,
                           const float* __restrict__ bnbr, const float* __restrict__ bnbi,
                           float* __restrict__ coef /* [C][6] */) {
  __shared__ float sm[256];
  const int c = blockIdx.x;
  const int tid = threadIdx.x;
  const long tot = (long)N * HW;
  float s1 = 0.f, s2 = 0.f, s3 = 0.f, s4 = 0.f, s5 = 0.f;
  for (long i = tid; i < tot; i += blockDim.x) {
    int n = (int)(i / HW);
    int p = (int)(i - (long)n * HW);
    float yr = conv[((long)n * 2 * C + c) * HW + p];
    float yi = conv[((long)n * 2 * C + C + c) * HW + p];
    s1 += yr; s2 += yi; s3 += yr * yr; s4 += yi * yi; s5 += yr * yi;
  }
  float vals[5] = {s1, s2, s3, s4, s5};
  float red[5];
#pragma unroll
  for (int q = 0; q < 5; ++q) {
    sm[tid] = vals[q]; __syncthreads();
    for (int s = 128; s > 0; s >>= 1) {
      if (tid < s) sm[tid] += sm[tid + s];
      __syncthreads();
    }
    red[q] = sm[0]; __syncthreads();
  }
  if (tid == 0) {
    float inv_n = 1.f / (float)tot;
    float mr = red[0] * inv_n, mi = red[1] * inv_n;
    float Crr = red[2] * inv_n - mr * mr + BN_EPS;
    float Cii = red[3] * inv_n - mi * mi + BN_EPS;
    float Cri = red[4] * inv_n - mr * mi;
    float s = sqrtf(Crr * Cii - Cri * Cri);
    float t = sqrtf(Crr + Cii + 2.f * s);
    float inv = 1.f / (s * t);
    float Rrr = (Cii + s) * inv, Rii = (Crr + s) * inv, Rri = -Cri * inv;
    float g0 = grr[c], g1 = gri[c], g2 = gii[c];
    float Arr = g0 * Rrr + g1 * Rri;
    float Ari = g0 * Rri + g1 * Rii;
    float Air = g1 * Rrr + g2 * Rri;
    float Aii = g1 * Rri + g2 * Rii;
    float Br = bnbr[c] - Arr * mr - Ari * mi;
    float Bi = bnbi[c] - Air * mr - Aii * mi;
    coef[c * 6 + 0] = Arr; coef[c * 6 + 1] = Ari; coef[c * 6 + 2] = Air;
    coef[c * 6 + 3] = Aii; coef[c * 6 + 4] = Br;  coef[c * 6 + 5] = Bi;
  }
}

// ---------------- fused BN-affine + CReLU + |z| 2x2 maxpool ----------------
__global__ void k_bn_relu_pool(const float* __restrict__ conv, const float* __restrict__ coef,
                               h16* __restrict__ outPool, int N, int C, int H, int W) {
  int idx = blockIdx.x * blockDim.x + threadIdx.x;
  int pH = H >> 1, pW = W >> 1;
  long tot = (long)N * C * pH * pW;
  if (idx >= tot) return;
  int px = idx % pW; int t = idx / pW;
  int py = t % pH;   t /= pH;
  int c = t % C;     int n = t / C;
  float Arr = coef[c * 6 + 0], Ari = coef[c * 6 + 1], Air = coef[c * 6 + 2];
  float Aii = coef[c * 6 + 3], Br = coef[c * 6 + 4],  Bi = coef[c * 6 + 5];
  float bestR = 0.f, bestI = 0.f, bestM = -1.f;
#pragma unroll
  for (int dy = 0; dy < 2; ++dy)
#pragma unroll
    for (int dx = 0; dx < 2; ++dx) {
      int y = 2 * py + dy, x = 2 * px + dx;
      float yr = conv[((long)n * 2 * C + c) * H * W + y * W + x];
      float yi = conv[((long)n * 2 * C + C + c) * H * W + y * W + x];
      float r  = fmaxf(Arr * yr + Ari * yi + Br, 0.f);
      float im = fmaxf(Air * yr + Aii * yi + Bi, 0.f);
      float m = r * r + im * im;
      if (m > bestM) { bestM = m; bestR = r; bestI = im; }   // first-max tie break
    }
  outPool[((long)n * 2 * C + c) * pH * pW + py * pW + px]     = (h16)bestR;
  outPool[((long)n * 2 * C + C + c) * pH * pW + py * pW + px] = (h16)bestI;
}

// ------------------------------ FC GEMM (WMMA) ------------------------------
// C[M x Ncols] = A[M x K] * Bsrc^T   (Bsrc row-major [Nbound][K])
// mode 1: fc1 -> complex bias + ReLU, scatter to er/ei (out0/out1, stride 256)
// mode 2: fc2 -> bias, store logits (out0, row stride Nbound)
__global__ void k_fc_wmma(const h16* __restrict__ A, const h16* __restrict__ Bsrc,
                          int M, int Nbound, int K, int KC, int mode,
                          const float* __restrict__ bv0, const float* __restrict__ bv1,
                          float* __restrict__ out0, float* __restrict__ out1) {
  extern __shared__ v16h ldsraw2[];
  h16* lds = (h16*)ldsraw2;
  const int KCpad = KC + 16;
  const int COLS = 16 * NB;
  const int tid = threadIdx.x;
  const int lane = tid & 31;
  const int wave = tid >> 5;
  const int col = lane & 15;
  const int khalf = lane >> 4;
  const int pBase = blockIdx.x * COLS;
  const int mTile = blockIdx.y * 8 + wave;
  const bool live = (mTile * 16) < M;

  v8f acc[NB] = {};
  for (int kc0 = 0; kc0 < K; kc0 += KC) {
    for (int i = tid; i < COLS * KC; i += blockDim.x) {
      int colG = i / KC;
      int kLoc = i - colG * KC;
      int k = kc0 + kLoc;
      int j = pBase + colG;
      h16 v = (h16)0.f;
      if (k < K && j < Nbound) v = Bsrc[(long)j * K + k];
      lds[colG * KCpad + kLoc] = v;
    }
    __syncthreads();
    if (live) {
      int kcLen = K - kc0; if (kcLen > KC) kcLen = KC;
      const h16* Arow = A + (long)(mTile * 16 + col) * K + kc0 + 16 * khalf;
      const h16* Bcol = lds + col * KCpad + 16 * khalf;
      for (int k0 = 0; k0 < kcLen; k0 += 32) {
        v8h alo = *(const v8h*)(Arow + k0);
        v8h ahi = *(const v8h*)(Arow + k0 + 8);
        v16h a;
#pragma unroll
        for (int t = 0; t < 8; ++t) { a[t] = alo[t]; a[8 + t] = ahi[t]; }
#pragma unroll
        for (int nb = 0; nb < NB; ++nb) {
          v16h b = *(const v16h*)(Bcol + nb * 16 * KCpad + k0);
          acc[nb] = __builtin_amdgcn_wmma_f32_16x16x32_f16(
              false, a, false, b, (short)0, acc[nb], false, false);
        }
      }
    }
    __syncthreads();
  }

  if (!live) return;
#pragma unroll
  for (int nb = 0; nb < NB; ++nb) {
    int j = pBase + nb * 16 + col;
#pragma unroll
    for (int v = 0; v < 8; ++v) {
      int row = mTile * 16 + v + 8 * khalf;   // batch index n
      float val = acc[nb][v];
      if (mode == 1) {
        float b = (j < 256) ? (bv0[j] - bv1[j]) : (bv0[j - 256] + bv1[j - 256]);
        val = fmaxf(val + b, 0.f);
        if (j < 256) out0[(long)row * 256 + j] = val;
        else         out1[(long)row * 256 + (j - 256)] = val;
      } else {
        if (j < Nbound) out0[(long)row * Nbound + j] = val + bv0[j];
      }
    }
  }
}

__global__ void k_mag(const float* __restrict__ er, const float* __restrict__ ei,
                      h16* __restrict__ mag, int n) {
  int i = blockIdx.x * blockDim.x + threadIdx.x;
  if (i < n) {
    float r = er[i], im = ei[i];
    mag[i] = (h16)sqrtf(r * r + im * im + MAG_EPS);
  }
}

// --------------------------------- launch ----------------------------------

extern "C" void kernel_launch(void* const* d_in, const int* in_sizes, int n_in,
                              void* d_out, int out_size, void* d_ws, size_t ws_size,
                              hipStream_t stream) {
  (void)in_sizes; (void)n_in; (void)out_size; (void)ws_size;
  // ---- inputs (setup_inputs dict order) ----
  const float* x = (const float*)d_in[0];
  const float* c_wr[4] = {(const float*)d_in[1],  (const float*)d_in[5],
                          (const float*)d_in[9],  (const float*)d_in[13]};
  const float* c_wi[4] = {(const float*)d_in[2],  (const float*)d_in[6],
                          (const float*)d_in[10], (const float*)d_in[14]};
  const float* c_br[4] = {(const float*)d_in[3],  (const float*)d_in[7],
                          (const float*)d_in[11], (const float*)d_in[15]};
  const float* c_bi[4] = {(const float*)d_in[4],  (const float*)d_in[8],
                          (const float*)d_in[12], (const float*)d_in[16]};
  const float* b_grr[4] = {(const float*)d_in[17], (const float*)d_in[22],
                           (const float*)d_in[27], (const float*)d_in[32]};
  const float* b_gri[4] = {(const float*)d_in[18], (const float*)d_in[23],
                           (const float*)d_in[28], (const float*)d_in[33]};
  const float* b_gii[4] = {(const float*)d_in[19], (const float*)d_in[24],
                           (const float*)d_in[29], (const float*)d_in[34]};
  const float* b_br[4]  = {(const float*)d_in[20], (const float*)d_in[25],
                           (const float*)d_in[30], (const float*)d_in[35]};
  const float* b_bi[4]  = {(const float*)d_in[21], (const float*)d_in[26],
                           (const float*)d_in[31], (const float*)d_in[36]};
  const float* fc1_wr = (const float*)d_in[37];
  const float* fc1_wi = (const float*)d_in[38];
  const float* fc1_br = (const float*)d_in[39];
  const float* fc1_bi = (const float*)d_in[40];
  const float* fc2_w  = (const float*)d_in[41];
  const float* fc2_b  = (const float*)d_in[42];

  float* outF  = (float*)d_out;
  float* out_logits = outF;                      // [512][1000]
  float* out_er = outF + (size_t)512 * 1000;     // [512][256]
  float* out_ei = out_er + (size_t)512 * 256;    // [512][256]

  // ---- workspace carve ----
  char* ws = (char*)d_ws;
  size_t off = 0;
  auto carve = [&](size_t bytes) -> char* {
    off = (off + 255) & ~(size_t)255;
    char* p = ws + off; off += bytes; return p;
  };
  h16*   xf16  = (h16*)carve((size_t)512 * 4096 * 2);
  float* convB = (float*)carve((size_t)512 * 32 * 4096 * 4);   // 256 MiB, reused
  h16*   poolA = (h16*)carve((size_t)512 * 32 * 1024 * 2);     // pool1 / pool3
  h16*   poolB = (h16*)carve((size_t)512 * 64 * 256 * 2);      // pool2 / pool4
  h16*   W1  = (h16*)carve((size_t)32 * 128 * 2);
  h16*   W2  = (h16*)carve((size_t)64 * 800 * 2);
  h16*   W3  = (h16*)carve((size_t)128 * 576 * 2);
  h16*   W4  = (h16*)carve((size_t)128 * 1152 * 2);
  h16*   Wf1 = (h16*)carve((size_t)512 * 2048 * 2);
  h16*   Wf2 = (h16*)carve((size_t)1000 * 256 * 2);
  float* biasAll = (float*)carve(352 * 4);    // 32 + 64 + 128 + 128
  float* coef    = (float*)carve(1056 * 4);   // (16+32+64+64)*6
  h16*   magB    = (h16*)carve((size_t)512 * 256 * 2);

  float* bias1 = biasAll;       float* bias2 = biasAll + 32;
  float* bias3 = biasAll + 96;  float* bias4 = biasAll + 224;
  float* coef1 = coef;          float* coef2 = coef + 96;
  float* coef3 = coef + 288;    float* coef4 = coef + 672;

  // ---- prep: f16 casts + packed complex weight blocks ----
  k_cast_f16<<<(2097152 + 255) / 256, 256, 0, stream>>>(x, xf16, 2097152);
  k_prep_conv1<<<16, 256, 0, stream>>>(c_wr[0], c_wi[0], c_br[0], c_bi[0], W1, bias1);
  k_prep_convc<<<(64 * 800 + 255) / 256, 256, 0, stream>>>(c_wr[1], c_wi[1], c_br[1], c_bi[1],
                                                           W2, bias2, 32, 16 * 25);
  k_prep_convc<<<(128 * 576 + 255) / 256, 256, 0, stream>>>(c_wr[2], c_wi[2], c_br[2], c_bi[2],
                                                            W3, bias3, 64, 32 * 9);
  k_prep_convc<<<(128 * 1152 + 255) / 256, 256, 0, stream>>>(c_wr[3], c_wi[3], c_br[3], c_bi[3],
                                                             W4, bias4, 64, 64 * 9);
  k_prep_fc1<<<(512 * 2048 + 255) / 256, 256, 0, stream>>>(fc1_wr, fc1_wi, Wf1);
  k_cast_f16<<<(256000 + 255) / 256, 256, 0, stream>>>(fc2_w, Wf2, 256000);

  auto launch_conv = [&](const h16* Wp, const float* bias, const h16* act, float* out,
                         int Cin, int inBlocks, int Hh, int Kk, int pad,
                         int Mrows, int K, int Kp, int KC) {
    int nTiles = (512 * Hh * Hh) / 16;          // always divisible by NB here
    int nBlocks = nTiles / NB;
    int waves = Mrows / 16; if (waves > 8) waves = 8;
    size_t lds = (size_t)NB * 16 * (KC + 16) * sizeof(h16);
    k_conv_wmma<<<dim3(nBlocks), dim3(waves * 32), lds, stream>>>(
        Wp, bias, act, out, 512, Cin, inBlocks, Hh, Hh, Kk, Kk, pad, Mrows, K, Kp, KC);
  };
  auto launch_bn_pool = [&](float* convOut, const float* grr, const float* gri,
                            const float* gii, const float* bbr, const float* bbi,
                            float* coefL, h16* poolOut, int C, int Hh) {
    k_bn_stats<<<C, 256, 0, stream>>>(convOut, 512, C, Hh * Hh, grr, gri, gii, bbr, bbi, coefL);
    long tot = (long)512 * C * (Hh / 2) * (Hh / 2);
    k_bn_relu_pool<<<(int)((tot + 255) / 256), 256, 0, stream>>>(convOut, coefL, poolOut,
                                                                 512, C, Hh, Hh);
  };

  // ---- layer 1: 1->16ch, 11x11, H=64 (M=32, K=121->128) ----
  launch_conv(W1, bias1, xf16, convB, 1, 1, 64, 11, 5, 32, 121, 128, 128);
  launch_bn_pool(convB, b_grr[0], b_gri[0], b_gii[0], b_br[0], b_bi[0], coef1, poolA, 16, 64);
  // ---- layer 2: 16->32ch, 5x5, H=32 (M=64, K=800) ----
  launch_conv(W2, bias2, poolA, convB, 16, 2, 32, 5, 2, 64, 800, 800, 256);
  launch_bn_pool(convB, b_grr[1], b_gri[1], b_gii[1], b_br[1], b_bi[1], coef2, poolB, 32, 32);
  // ---- layer 3: 32->64ch, 3x3, H=16 (M=128, K=576) ----
  launch_conv(W3, bias3, poolB, convB, 32, 2, 16, 3, 1, 128, 576, 576, 256);
  launch_bn_pool(convB, b_grr[2], b_gri[2], b_gii[2], b_br[2], b_bi[2], coef3, poolA, 64, 16);
  // ---- layer 4: 64->64ch, 3x3, H=8 (M=128, K=1152) ----
  launch_conv(W4, bias4, poolA, convB, 64, 2, 8, 3, 1, 128, 1152, 1152, 256);
  launch_bn_pool(convB, b_grr[3], b_gri[3], b_gii[3], b_br[3], b_bi[3], coef4, poolB, 64, 8);

  // ---- fc1: [512 x 2048] x [2048 x 512] -> er/ei (+bias, ReLU) ----
  {
    size_t lds = (size_t)NB * 16 * (256 + 16) * sizeof(h16);
    k_fc_wmma<<<dim3(512 / (16 * NB), 4), 256, lds, stream>>>(
        poolB, Wf1, 512, 512, 2048, 256, 1, fc1_br, fc1_bi, out_er, out_ei);
  }
  // ---- magnitude ----
  k_mag<<<(131072 + 255) / 256, 256, 0, stream>>>(out_er, out_ei, magB, 131072);
  // ---- fc2: [512 x 256] x [256 x 1000] -> logits ----
  {
    size_t lds = (size_t)NB * 16 * (256 + 16) * sizeof(h16);
    int nBlocks = (1000 + 16 * NB - 1) / (16 * NB);   // 16 blocks, tail guarded
    k_fc_wmma<<<dim3(nBlocks, 4), 256, lds, stream>>>(
        magB, Wf2, 512, 1000, 256, 256, 2, fc2_b, nullptr, out_logits, nullptr);
  }
}
